// MultiHeadAttentionLayer_36077725286491
// MI455X (gfx1250) — compile-verified, matrix operand order checked by reading, not run on hardware
//
#include <hip/hip_runtime.h>

// ===========================================================================
// Windowed cross-attention (Swin-style, qk_ratio=2) for MI455X / gfx1250.
// bf16 WMMA everywhere; async global->LDS staging in the attention kernel.
// ===========================================================================

#define USE_ASYNC 1

typedef __attribute__((ext_vector_type(16))) __bf16 v16bf;
typedef __attribute__((ext_vector_type(8)))  float  v8f;

// ---- workspace layout (bytes) ------------------------------------------------
#define OFF_WQT   0u                       // WqT  [256n][256k] bf16   (128 KB)
#define OFF_WKVT  131072u                  // WkvT [512n][256k] bf16   (256 KB)
#define OFF_WOT   393216u                  // WoT  [256n][256k] bf16   (128 KB)
#define OFF_QW    524288u                  // q windows [b][h][win][256][32] bf16
#define OFF_KW    (524288u + 67108864u)    // k windows [b][h][win][64][32] bf16
#define OFF_VW    (OFF_KW + 16777216u)     // v windows [b][h][win][32][64] bf16 (transposed!)
#define OFF_OW    (OFF_VW + 16777216u)     // attn out  [b][h][win][256][32] bf16

// ---- small helpers -----------------------------------------------------------
__device__ __forceinline__ unsigned short f2bf(float f){
  unsigned int x = __builtin_bit_cast(unsigned int, f);
  x += 0x7fffu + ((x >> 16) & 1u);               // round-to-nearest-even
  return (unsigned short)(x >> 16);
}
__device__ __forceinline__ unsigned int pk2(float a, float b){
  return (unsigned int)f2bf(a) | ((unsigned int)f2bf(b) << 16);
}
__device__ __forceinline__ v8f mk0(){ v8f z = {0.f,0.f,0.f,0.f,0.f,0.f,0.f,0.f}; return z; }

// Load a 16x32 bf16 A-fragment (or N-major B-fragment) from LDS.
// base points at element K=0 of this lane's row (A: row = tile_m + (lane&15);
// B stored [N][K]: base = &B[tile_n + (lane&15)][0]).  K pairs are contiguous.
__device__ __forceinline__ v16bf frag_ld(const unsigned short* base, int lane){
  union { unsigned int w[8]; v16bf v; } u;
  const int kb = (lane & 16) >> 1;               // +8 for upper lane half
  #pragma unroll
  for (int j = 0; j < 8; ++j){
    const int k = ((j >> 2) << 4) + kb + ((j & 3) << 1);
    u.w[j] = *reinterpret_cast<const unsigned int*>(base + k);
  }
  return u.v;
}

__device__ __forceinline__ v8f wmma_bf16(v16bf a, v16bf b, v8f c){
  return __builtin_amdgcn_wmma_f32_16x16x32_bf16(false, a, false, b, (short)0, c,
                                                 false, false);
}

#if USE_ASYNC
__device__ __forceinline__ unsigned lds_addr32(const void* p){
  return (unsigned)(unsigned long long)p;        // low 32 bits = LDS offset
}
__device__ __forceinline__ void async_b128(unsigned loff, const void* g){
  asm volatile("global_load_async_to_lds_b128 %0, %1, off"
               :: "v"(loff), "v"(g) : "memory");
}
__device__ __forceinline__ void wait_async0(){
  asm volatile("s_wait_asynccnt 0x0" ::: "memory");
}
#endif

// ===========================================================================
// Kernel 1: convert weights fp32 -> bf16, transposed to [N][K]
// ===========================================================================
__global__ void __launch_bounds__(256) k_convert(
    const float* __restrict__ Wq, const float* __restrict__ Wkv,
    const float* __restrict__ Wo,
    unsigned short* __restrict__ wqT, unsigned short* __restrict__ wkvT,
    unsigned short* __restrict__ woT)
{
  const int id = blockIdx.x * 256 + threadIdx.x;
  if (id < 65536){                 // Wq 256x256
    const int n = id >> 8, k = id & 255;
    wqT[n * 256 + k] = f2bf(Wq[k * 256 + n]);
  } else if (id < 196608){         // Wkv 256x512
    const int l = id - 65536, n = l >> 8, k = l & 255;
    wkvT[n * 256 + k] = f2bf(Wkv[k * 512 + n]);
  } else if (id < 262144){         // Wo 256x256
    const int l = id - 196608, n = l >> 8, k = l & 255;
    woT[n * 256 + k] = f2bf(Wo[k * 256 + n]);
  }
}

// ===========================================================================
// Kernel 2: q projection, fused roll(-8,-8) + 16x16 window partition
// block = (b, win): 256 tokens x 256 dims, K=256.  512 thr = 16 waves (4x4),
// each wave owns a 64x64 tile region (16 16x16 WMMA accumulators).
// ===========================================================================
__global__ void __launch_bounds__(512) k_qproj(
    const float* __restrict__ x, const unsigned short* __restrict__ wqT,
    const float* __restrict__ bq, unsigned short* __restrict__ qw)
{
  __shared__ __align__(16) unsigned short As[256 * 36];  // tokens x Kc(32), pad->36
  __shared__ __align__(16) unsigned short Bs[256 * 36];  // N x Kc(32)
  const int bid = blockIdx.x, win = bid & 63, b = bid >> 6;
  const int wy = win >> 3, wx = win & 7;
  const int tid = threadIdx.x, lane = tid & 31, wid = tid >> 5;
  const int wm = wid >> 2, wn = wid & 3;
  const int ln = lane & 15, hi2 = (lane >> 4) << 3;

  const int t = tid >> 1, half = tid & 1;                // staging role
  const int ti = t >> 4, tj = t & 15;
  const int hh = (wy * 16 + ti + 8) & 127, wwc = (wx * 16 + tj + 8) & 127;
  const float* xrow = x + ((((b << 7) + hh) << 7) + wwc) * 256;

  v8f acc[4][4];
  #pragma unroll
  for (int i = 0; i < 4; ++i)
    #pragma unroll
    for (int j = 0; j < 4; ++j) acc[i][j] = mk0();

  for (int kc = 0; kc < 8; ++kc){
    const int K0 = kc << 5;
    __syncthreads();
    { // stage A: 16 fp32 -> bf16 per thread
      const float4* p = (const float4*)(xrow + K0 + (half << 4));
      unsigned int* d = (unsigned int*)&As[t * 36 + (half << 4)];
      #pragma unroll
      for (int i = 0; i < 4; ++i){
        float4 v = p[i];
        d[2 * i]     = pk2(v.x, v.y);
        d[2 * i + 1] = pk2(v.z, v.w);
      }
    }
    { // stage B: 16 bf16 per thread from WqT[n][K0..]
      const uint4* p = (const uint4*)(wqT + t * 256 + K0 + (half << 4));
      uint4 u0 = p[0], u1 = p[1];
      unsigned int* d = (unsigned int*)&Bs[t * 36 + (half << 4)];
      d[0] = u0.x; d[1] = u0.y; d[2] = u0.z; d[3] = u0.w;
      d[4] = u1.x; d[5] = u1.y; d[6] = u1.z; d[7] = u1.w;
    }
    __syncthreads();
    v16bf bf[4];
    #pragma unroll
    for (int nt = 0; nt < 4; ++nt)
      bf[nt] = frag_ld(&Bs[((wn << 6) + (nt << 4) + ln) * 36], lane);
    #pragma unroll
    for (int mt = 0; mt < 4; ++mt){
      v16bf af = frag_ld(&As[((wm << 6) + (mt << 4) + ln) * 36], lane);
      #pragma unroll
      for (int nt = 0; nt < 4; ++nt)
        acc[mt][nt] = wmma_bf16(af, bf[nt], acc[mt][nt]);
    }
  }
  // epilogue: +bq, -> bf16 per-head windows
  #pragma unroll
  for (int mt = 0; mt < 4; ++mt)
    #pragma unroll
    for (int nt = 0; nt < 4; ++nt){
      const int N = (wn << 6) + (nt << 4) + ln;
      const int headN = N >> 5, dk = N & 31;
      const float bias = bq[N];
      const int base = ((((b << 3) + headN) << 6) + win) * 8192 + dk;
      #pragma unroll
      for (int r = 0; r < 8; ++r){
        const int M = (wm << 6) + (mt << 4) + r + hi2;
        qw[base + (M << 5)] = f2bf(acc[mt][nt][r] + bias);
      }
    }
}

// ===========================================================================
// Kernel 3: kv projection, fused roll(-4,-4) + 8x8 partition.
// block = (b, win): 64 tokens x 512 dims.  16 waves (2x8), 32x64 per wave.
// k stored [tok][dk]; v stored transposed [dk][tok] for the P@V B-fragments.
// ===========================================================================
__global__ void __launch_bounds__(512) k_kvproj(
    const float* __restrict__ z, const unsigned short* __restrict__ wkvT,
    const float* __restrict__ bkv, unsigned short* __restrict__ kw,
    unsigned short* __restrict__ vw)
{
  __shared__ __align__(16) unsigned short As[64 * 36];
  __shared__ __align__(16) unsigned short Bs[512 * 36];
  const int bid = blockIdx.x, win = bid & 63, b = bid >> 6;
  const int wy = win >> 3, wx = win & 7;
  const int tid = threadIdx.x, lane = tid & 31, wid = tid >> 5;
  const int wm = wid >> 3, wn = wid & 7;
  const int ln = lane & 15, hi2 = (lane >> 4) << 3;

  const int t = tid >> 3, c0 = (tid & 7) << 2;           // A staging role
  const int ti = t >> 3, tj = t & 7;
  const int hh = (wy * 8 + ti + 4) & 63, wwc = (wx * 8 + tj + 4) & 63;
  const float* zrow = z + ((((b << 6) + hh) << 6) + wwc) * 256;

  v8f acc[2][4];
  #pragma unroll
  for (int i = 0; i < 2; ++i)
    #pragma unroll
    for (int j = 0; j < 4; ++j) acc[i][j] = mk0();

  for (int kc = 0; kc < 8; ++kc){
    const int K0 = kc << 5;
    __syncthreads();
    { const float4 v = *(const float4*)(zrow + K0 + c0);
      unsigned int* d = (unsigned int*)&As[t * 36 + c0];
      d[0] = pk2(v.x, v.y); d[1] = pk2(v.z, v.w);
    }
    { const uint4* p = (const uint4*)(wkvT + tid * 256 + K0);
      uint4 u0 = p[0], u1 = p[1], u2 = p[2], u3 = p[3];
      unsigned int* d = (unsigned int*)&Bs[tid * 36];
      d[0]=u0.x;  d[1]=u0.y;  d[2]=u0.z;  d[3]=u0.w;
      d[4]=u1.x;  d[5]=u1.y;  d[6]=u1.z;  d[7]=u1.w;
      d[8]=u2.x;  d[9]=u2.y;  d[10]=u2.z; d[11]=u2.w;
      d[12]=u3.x; d[13]=u3.y; d[14]=u3.z; d[15]=u3.w;
    }
    __syncthreads();
    v16bf bf[4];
    #pragma unroll
    for (int nt = 0; nt < 4; ++nt)
      bf[nt] = frag_ld(&Bs[((wn << 6) + (nt << 4) + ln) * 36], lane);
    #pragma unroll
    for (int mt = 0; mt < 2; ++mt){
      v16bf af = frag_ld(&As[((wm << 5) + (mt << 4) + ln) * 36], lane);
      #pragma unroll
      for (int nt = 0; nt < 4; ++nt)
        acc[mt][nt] = wmma_bf16(af, bf[nt], acc[mt][nt]);
    }
  }
  #pragma unroll
  for (int mt = 0; mt < 2; ++mt)
    #pragma unroll
    for (int nt = 0; nt < 4; ++nt){
      const int N = (wn << 6) + (nt << 4) + ln;       // 0..511
      const int pair = N >> 8, headN = (N >> 5) & 7, dk = N & 31;
      const float bias = bkv[N];
      const int wbase = ((((b << 3) + headN) << 6) + win) * 2048;
      #pragma unroll
      for (int r = 0; r < 8; ++r){
        const int M = (wm << 5) + (mt << 4) + r + hi2;
        const unsigned short vbf = f2bf(acc[mt][nt][r] + bias);
        if (pair == 0) kw[wbase + (M << 5) + dk] = vbf;       // k: [tok][dk]
        else           vw[wbase + (dk << 6) + M] = vbf;       // v: [dk][tok]
      }
    }
}

// ===========================================================================
// Kernel 4: attention per (b, head, win).  256 thr = 8 waves; wave owns 32 q rows.
// Async global->LDS staging of q/k/v tiles; S and O via bf16 WMMA;
// softmax rows live entirely inside one lane-half (shfl_xor width 16).
// ===========================================================================
__global__ void __launch_bounds__(256) k_attn(
    const unsigned short* __restrict__ qw, const unsigned short* __restrict__ kw,
    const unsigned short* __restrict__ vw, const unsigned char* __restrict__ mask,
    const float* __restrict__ rel, unsigned short* __restrict__ ow)
{
  __shared__ __align__(16) unsigned char smem[41472];
  unsigned short* vs = (unsigned short*)(smem);            // [32][72]  v (persist)
  unsigned short* qs = (unsigned short*)(smem + 4608);     // [256][40] phase 1
  unsigned short* ks = (unsigned short*)(smem + 25088);    // [64][40]  phase 1
  unsigned short* ps = (unsigned short*)(smem + 4608);     // [256][72] phase 2

  const int bid = blockIdx.x;
  const int win = bid & 63, head = (bid >> 6) & 7, b = bid >> 9;
  const int wy = win >> 3, wx = win & 7;
  const int tid = threadIdx.x, lane = tid & 31, wv = tid >> 5;
  const int ln = lane & 15, hi2 = (lane >> 4) << 3;

  const int bhw = (b * 8 + head) * 64 + win;
  const unsigned short* qg = qw + ((size_t)bhw << 13);     // 256*32
  const unsigned short* kg = kw + ((size_t)bhw << 11);     // 64*32
  const unsigned short* vg = vw + ((size_t)bhw << 11);     // 32*64

#if USE_ASYNC
  #pragma unroll
  for (int it = 0; it < 4; ++it){
    const int li = it * 256 + tid, row = li >> 2, seg = li & 3;
    async_b128(lds_addr32(qs) + row * 80u + seg * 16u,
               (const char*)qg + row * 64 + seg * 16);
  }
  { const int row = tid >> 2, seg = tid & 3;
    async_b128(lds_addr32(ks) + row * 80u + seg * 16u,
               (const char*)kg + row * 64 + seg * 16); }
  { const int dk = tid >> 3, seg = tid & 7;
    async_b128(lds_addr32(vs) + dk * 144u + seg * 16u,
               (const char*)vg + dk * 128 + seg * 16); }
  wait_async0();
#else
  #pragma unroll
  for (int it = 0; it < 4; ++it){
    const int li = it * 256 + tid, row = li >> 2, seg = li & 3;
    *(uint4*)&qs[row * 40 + seg * 8] = *(const uint4*)(qg + row * 32 + seg * 8);
  }
  { const int row = tid >> 2, seg = tid & 3;
    *(uint4*)&ks[row * 40 + seg * 8] = *(const uint4*)(kg + row * 32 + seg * 8); }
  { const int dk = tid >> 3, seg = tid & 7;
    *(uint4*)&vs[dk * 72 + seg * 8] = *(const uint4*)(vg + dk * 64 + seg * 8); }
#endif
  __syncthreads();

  // ---- S = q @ k^T  (M=32 rows/wave, N=64 keys, K=32) ----
  const int Mb = wv << 5;
  v8f sacc[2][4];
  #pragma unroll
  for (int mt = 0; mt < 2; ++mt)
    #pragma unroll
    for (int nt = 0; nt < 4; ++nt) sacc[mt][nt] = mk0();
  v16bf kb4[4];
  #pragma unroll
  for (int nt = 0; nt < 4; ++nt)
    kb4[nt] = frag_ld(&ks[((nt << 4) + ln) * 40], lane);
  #pragma unroll
  for (int mt = 0; mt < 2; ++mt){
    v16bf qa = frag_ld(&qs[(Mb + (mt << 4) + ln) * 40], lane);
    #pragma unroll
    for (int nt = 0; nt < 4; ++nt)
      sacc[mt][nt] = wmma_bf16(qa, kb4[nt], sacc[mt][nt]);
  }
  __syncthreads();   // all qs/ks reads done before ps aliases the region

  // ---- scale + rel-bias + mask + softmax; write P (bf16) to LDS ----
  const unsigned char* mp = mask + ((size_t)(((head * 8 + wy) * 8) + wx) << 14);
  const float SCALE = 0.17677669529663687f;  // 1/sqrt(32)
  #pragma unroll
  for (int mt = 0; mt < 2; ++mt){
    #pragma unroll
    for (int r = 0; r < 8; ++r){
      const int qt = Mb + (mt << 4) + r + hi2;
      const int ti = qt >> 4, tj = qt & 15;
      float vals[4];
      float vmax = -3.4e38f;
      #pragma unroll
      for (int nt = 0; nt < 4; ++nt){
        const int kt = (nt << 4) + ln;
        const int si = kt >> 3, sj = kt & 7;
        float s = sacc[mt][nt][r] * SCALE
                + rel[((ti - 2 * si + 14) * 30 + (tj - 2 * sj + 14)) * 8 + head];
        if (mp[qt * 64 + kt]) s = -1e9f;
        vals[nt] = s;
        vmax = fmaxf(vmax, s);
      }
      #pragma unroll
      for (int d = 1; d < 16; d <<= 1) vmax = fmaxf(vmax, __shfl_xor(vmax, d, 16));
      float sum = 0.f;
      #pragma unroll
      for (int nt = 0; nt < 4; ++nt){
        float p = __expf(vals[nt] - vmax); vals[nt] = p; sum += p;
      }
      #pragma unroll
      for (int d = 1; d < 16; d <<= 1) sum += __shfl_xor(sum, d, 16);
      const float is = 1.f / sum;
      #pragma unroll
      for (int nt = 0; nt < 4; ++nt)
        ps[qt * 72 + (nt << 4) + ln] = f2bf(vals[nt] * is);
    }
  }
  __syncthreads();

  // ---- O = P @ V  (M=32, N=32, K=64 in two K-chunks) ----
  v8f oacc[2][2];
  #pragma unroll
  for (int mt = 0; mt < 2; ++mt)
    #pragma unroll
    for (int nv = 0; nv < 2; ++nv) oacc[mt][nv] = mk0();
  #pragma unroll
  for (int kc = 0; kc < 2; ++kc){
    v16bf vb[2];
    #pragma unroll
    for (int nv = 0; nv < 2; ++nv)
      vb[nv] = frag_ld(&vs[((nv << 4) + ln) * 72 + (kc << 5)], lane);
    #pragma unroll
    for (int mt = 0; mt < 2; ++mt){
      v16bf pa = frag_ld(&ps[(Mb + (mt << 4) + ln) * 72 + (kc << 5)], lane);
      #pragma unroll
      for (int nv = 0; nv < 2; ++nv)
        oacc[mt][nv] = wmma_bf16(pa, vb[nv], oacc[mt][nv]);
    }
  }
  unsigned short* og = ow + ((size_t)bhw << 13);
  #pragma unroll
  for (int mt = 0; mt < 2; ++mt)
    #pragma unroll
    for (int nv = 0; nv < 2; ++nv){
      const int dk = (nv << 4) + ln;
      #pragma unroll
      for (int r = 0; r < 8; ++r){
        const int qt = Mb + (mt << 4) + r + hi2;
        og[(qt << 5) + dk] = f2bf(oacc[mt][nv][r]);
      }
    }
}

// ===========================================================================
// Kernel 5: output projection.  block = (b, win); K-chunk kc == head, so the
// windowed O layout is gathered contiguously; scatter undoes partition+roll.
// ===========================================================================
__global__ void __launch_bounds__(512) k_oproj(
    const unsigned short* __restrict__ ow, const unsigned short* __restrict__ woT,
    const float* __restrict__ bo, float* __restrict__ out)
{
  __shared__ __align__(16) unsigned short As[256 * 36];
  __shared__ __align__(16) unsigned short Bs[256 * 36];
  const int bid = blockIdx.x, win = bid & 63, b = bid >> 6;
  const int wy = win >> 3, wx = win & 7;
  const int tid = threadIdx.x, lane = tid & 31, wid = tid >> 5;
  const int wm = wid >> 2, wn = wid & 3;
  const int ln = lane & 15, hi2 = (lane >> 4) << 3;
  const int t = tid >> 1, half = tid & 1;

  v8f acc[4][4];
  #pragma unroll
  for (int i = 0; i < 4; ++i)
    #pragma unroll
    for (int j = 0; j < 4; ++j) acc[i][j] = mk0();

  for (int kc = 0; kc < 8; ++kc){
    const int K0 = kc << 5;
    __syncthreads();
    { const uint4* p = (const uint4*)(ow + ((((b << 3) + kc) << 6) + win) * 8192
                                      + (t << 5) + (half << 4));
      uint4 u0 = p[0], u1 = p[1];
      unsigned int* d = (unsigned int*)&As[t * 36 + (half << 4)];
      d[0]=u0.x; d[1]=u0.y; d[2]=u0.z; d[3]=u0.w;
      d[4]=u1.x; d[5]=u1.y; d[6]=u1.z; d[7]=u1.w;
    }
    { const uint4* p = (const uint4*)(woT + t * 256 + K0 + (half << 4));
      uint4 u0 = p[0], u1 = p[1];
      unsigned int* d = (unsigned int*)&Bs[t * 36 + (half << 4)];
      d[0]=u0.x; d[1]=u0.y; d[2]=u0.z; d[3]=u0.w;
      d[4]=u1.x; d[5]=u1.y; d[6]=u1.z; d[7]=u1.w;
    }
    __syncthreads();
    v16bf bf[4];
    #pragma unroll
    for (int nt = 0; nt < 4; ++nt)
      bf[nt] = frag_ld(&Bs[((wn << 6) + (nt << 4) + ln) * 36], lane);
    #pragma unroll
    for (int mt = 0; mt < 4; ++mt){
      v16bf af = frag_ld(&As[((wm << 6) + (mt << 4) + ln) * 36], lane);
      #pragma unroll
      for (int nt = 0; nt < 4; ++nt)
        acc[mt][nt] = wmma_bf16(af, bf[nt], acc[mt][nt]);
    }
  }
  #pragma unroll
  for (int mt = 0; mt < 4; ++mt)
    #pragma unroll
    for (int nt = 0; nt < 4; ++nt){
      const int N = (wn << 6) + (nt << 4) + ln;
      const float bias = bo[N];
      #pragma unroll
      for (int r = 0; r < 8; ++r){
        const int M = (wm << 6) + (mt << 4) + r + hi2;
        const int mi = M >> 4, mj = M & 15;
        const int hh = (wy * 16 + mi + 8) & 127, wwc = (wx * 16 + mj + 8) & 127;
        out[((((b << 7) + hh) << 7) + wwc) * 256 + N] = acc[mt][nt][r] + bias;
      }
    }
}

// ===========================================================================
extern "C" void kernel_launch(void* const* d_in, const int* in_sizes, int n_in,
                              void* d_out, int out_size, void* d_ws, size_t ws_size,
                              hipStream_t stream)
{
  const float* x            = (const float*)d_in[0];
  const float* z            = (const float*)d_in[1];
  const unsigned char* mask = (const unsigned char*)d_in[2];
  const float* Wq           = (const float*)d_in[3];
  const float* bq           = (const float*)d_in[4];
  const float* Wkv          = (const float*)d_in[5];
  const float* bkv          = (const float*)d_in[6];
  const float* Wo           = (const float*)d_in[7];
  const float* bo           = (const float*)d_in[8];
  const float* rel          = (const float*)d_in[9];
  float* out = (float*)d_out;
  char* ws = (char*)d_ws;
  (void)in_sizes; (void)n_in; (void)out_size; (void)ws_size;

  unsigned short* wqT  = (unsigned short*)(ws + OFF_WQT);
  unsigned short* wkvT = (unsigned short*)(ws + OFF_WKVT);
  unsigned short* woT  = (unsigned short*)(ws + OFF_WOT);
  unsigned short* qw   = (unsigned short*)(ws + OFF_QW);
  unsigned short* kw   = (unsigned short*)(ws + OFF_KW);
  unsigned short* vw   = (unsigned short*)(ws + OFF_VW);
  unsigned short* ow   = (unsigned short*)(ws + OFF_OW);

  k_convert<<<1024, 256, 0, stream>>>(Wq, Wkv, Wo, wqT, wkvT, woT);
  k_qproj  <<<512, 512, 0, stream>>>(x, wqT, bq, qw);
  k_kvproj <<<512, 512, 0, stream>>>(z, wkvT, bkv, kw, vw);
  k_attn   <<<4096, 256, 0, stream>>>(qw, kw, vw, mask, rel, ow);
  k_oproj  <<<512, 512, 0, stream>>>(ow, woT, bo, out);
}
